// HeteroGNN_19920058318954
// MI455X (gfx1250) — matrix operand order repaired and checked by reading, not compiled, not run on hardware
//
#include <hip/hip_runtime.h>
#include <math.h>

typedef __attribute__((ext_vector_type(16))) _Float16 v16h;
typedef __attribute__((ext_vector_type(4)))  _Float16 v4h;
typedef __attribute__((ext_vector_type(8)))  float    v8f;

#define NNODE 131072
#define FDIM  64
#define NEDGE 1000000

__device__ __forceinline__ float gelu_f(float x) {
    return 0.5f * x * (1.0f + erff(x * 0.70710678118654752f));
}

// Y[r][n] = sum_k X[r][k] * W[k][n] (+ bias)
// mode 0: plain. mode 1: GELU applied to X elements on load; epilogue
//   Y = sigmoid(*skipGate) * (acc + bias) + (1 - sigmoid) * skipX
//
// Tiles are staged in LDS pre-converted to f16 in fragment-linear order, so
// each lane's v16h A/B fragment is one contiguous 32-byte LDS read.
__global__ __launch_bounds__(256)
void wmma_gemm64(const float* __restrict__ X, const float* __restrict__ W,
                 const float* __restrict__ bias, float* __restrict__ Y,
                 const float* __restrict__ skipX, const float* __restrict__ skipGate,
                 int mode)
{
    // B fragments: [kc][t][laneSlot][i], laneSlot = (n&15) + 16*khalf, i = k%16
    __shared__ __align__(32) _Float16 sBh[2][4][32][16];   // 8 KB
    // A fragments: [wave][kc][laneSlot][i], laneSlot = m + 16*hal
    __shared__ __align__(32) _Float16 sAh[8][2][32][16];   // 16 KB
    __shared__ float sB[64];

    const int tid  = threadIdx.x;
    const int lane = tid & 31;
    const int wave = tid >> 5;

    // ---- stage B (W 64x64 f32 -> f16 fragments) ----
    {
        int n  = tid & 63;           // output column
        int kb = tid >> 6;           // 16-row k block, 0..3
        int kc = kb >> 1, khalf = kb & 1;
        int t  = n >> 4, slot = (n & 15) + 16 * khalf;
        v16h hv;
        #pragma unroll
        for (int i = 0; i < 16; ++i)
            hv[i] = (_Float16)W[(kb * 16 + i) * 64 + n];   // coalesced across tid per i
        *(v16h*)&sBh[kc][t][slot][0] = hv;
    }
    if (tid < 64) sB[tid] = bias ? bias[tid] : 0.0f;

    // ---- stage A (this wave's 16x64 tile, optional fused GELU) ----
    const long rowBase = (long)blockIdx.x * 128 + wave * 16;
    const float4* Xv4  = (const float4*)(X + rowBase * FDIM);
    #pragma unroll
    for (int j = 0; j < 8; ++j) {
        int f  = lane + 32 * j;          // float4 index within the 16x64 tile
        int m  = f >> 4;                 // row within tile
        int k  = (f & 15) * 4;           // first k of this float4
        int kc = k >> 5;
        int kl = k & 31;
        int hal = kl >> 4;                           // which half-wave holds it
        int i   = kl & 15;                           // position within v16h
        float4 tv = Xv4[f];
        if (mode == 1) {
            tv.x = gelu_f(tv.x); tv.y = gelu_f(tv.y);
            tv.z = gelu_f(tv.z); tv.w = gelu_f(tv.w);
        }
        v4h hv;
        hv[0] = (_Float16)tv.x; hv[1] = (_Float16)tv.y;
        hv[2] = (_Float16)tv.z; hv[3] = (_Float16)tv.w;
        *(v4h*)&sAh[wave][kc][m + 16 * hal][i] = hv;
    }
    // Speculative prefetch of the next row block (-> global_prefetch_b8).
    __builtin_prefetch(X + (rowBase + 128) * FDIM + lane * 4, 0, 1);

    __syncthreads();

    // ---- WMMA main loop: fragments are contiguous per lane ----
    v8f acc[4] = {};
    #pragma unroll
    for (int kc = 0; kc < 2; ++kc) {
        v16h a = *(const v16h*)&sAh[wave][kc][lane][0];
        #pragma unroll
        for (int t = 0; t < 4; ++t) {
            v16h b = *(const v16h*)&sBh[kc][t][lane][0];
            acc[t] = __builtin_amdgcn_wmma_f32_16x16x32_f16(
                false, a, false, b, (short)0, acc[t], false, false);
        }
    }

    // ---- epilogue ----
    float gate = 0.f, onemg = 0.f;
    if (mode == 1) {
        float s = *skipGate;
        gate  = 1.0f / (1.0f + __expf(-s));
        onemg = 1.0f - gate;
    }
    const int m    = lane & 15;
    const int half = lane >> 4;
    // C/D layout: VGPR v, lanes0-15 -> M=v, N=lane; lanes16-31 -> M=v+8.
    #pragma unroll
    for (int t = 0; t < 4; ++t)
        #pragma unroll
        for (int v = 0; v < 8; ++v) {
            int  mrow = v + 8 * half;
            int  col  = t * 16 + m;
            long row  = rowBase + mrow;
            float val = acc[t][v] + sB[col];
            if (mode == 1) val = gate * val + onemg * skipX[row * FDIM + col];
            Y[row * FDIM + col] = val;
        }
}

// ---- edge phase ------------------------------------------------------------

__device__ __forceinline__ unsigned f32_order_encode(float f) {
    unsigned b = __float_as_uint(f);
    return (b & 0x80000000u) ? ~b : (b | 0x80000000u);
}
__device__ __forceinline__ float f32_order_decode(unsigned k) {
    return __uint_as_float((k & 0x80000000u) ? (k ^ 0x80000000u) : ~k);
}

__global__ void init_mx_kernel(unsigned* mx, int n) {
    int i = blockIdx.x * blockDim.x + threadIdx.x;
    if (i < n) mx[i] = 0x007FFFFFu;   // encode(-inf)
}

// one wave per edge: score = (q[dst] . kr[src]) * p_rel / 8, running max per dst
__global__ __launch_bounds__(256)
void score_max_kernel(const float* __restrict__ q, const float* __restrict__ kr,
                      const int* __restrict__ src, const int* __restrict__ dst,
                      const float* __restrict__ prel, float* __restrict__ score,
                      unsigned* __restrict__ mxbits)
{
    int lane = threadIdx.x & 31;
    int e    = blockIdx.x * 8 + (threadIdx.x >> 5);
    long s = src[e], d = dst[e];
    float p = q[d * FDIM + lane]      * kr[s * FDIM + lane]
            + q[d * FDIM + 32 + lane] * kr[s * FDIM + 32 + lane];
    #pragma unroll
    for (int off = 16; off > 0; off >>= 1) p += __shfl_down(p, off, 32);
    if (lane == 0) {
        float sc = p * prel[0] * 0.125f;   // 1/sqrt(64)
        score[e] = sc;
        atomicMax(&mxbits[d], f32_order_encode(sc));
    }
}

__global__ void exp_sum_kernel(float* __restrict__ score, const int* __restrict__ dst,
                               const unsigned* __restrict__ mxbits, float* __restrict__ den)
{
    int e = blockIdx.x * blockDim.x + threadIdx.x;
    if (e >= NEDGE) return;
    int d = dst[e];
    float ex = __expf(score[e] - f32_order_decode(mxbits[d]));
    score[e] = ex;
    atomicAdd(&den[d], ex);
}

// one wave per edge: agg[dst] += (ex/den[dst]) * vr[src]
__global__ __launch_bounds__(256)
void scatter_kernel(const float* __restrict__ score, const float* __restrict__ den,
                    const float* __restrict__ vr, const int* __restrict__ src,
                    const int* __restrict__ dst, float* __restrict__ agg)
{
    int lane = threadIdx.x & 31;
    int e    = blockIdx.x * 8 + (threadIdx.x >> 5);
    long s = src[e], d = dst[e];
    float alpha = score[e] / den[d];
    atomicAdd(&agg[d * FDIM + lane],      alpha * vr[s * FDIM + lane]);
    atomicAdd(&agg[d * FDIM + 32 + lane], alpha * vr[s * FDIM + 32 + lane]);
}

// ---- readout ---------------------------------------------------------------

__global__ void pool_kernel(const float* __restrict__ x, float* __restrict__ pooled) {
    int  col = threadIdx.x & 63;
    long row = (long)blockIdx.x * 4 + (threadIdx.x >> 6);
    long stride = (long)gridDim.x * 4;
    float s = 0.f;
    for (; row < NNODE; row += stride) s += x[row * FDIM + col];
    atomicAdd(&pooled[col], s);
}

__global__ void final_kernel(const float* __restrict__ pooled, const float* __restrict__ lw,
                             const float* __restrict__ lb, float* __restrict__ out)
{
    int lane = threadIdx.x;
    float p = pooled[lane] * lw[lane] + pooled[lane + 32] * lw[lane + 32];
    #pragma unroll
    for (int off = 16; off > 0; off >>= 1) p += __shfl_down(p, off, 32);
    if (lane == 0) out[0] = p + lb[0];
}

// ---- host ------------------------------------------------------------------

extern "C" void kernel_launch(void* const* d_in, const int* in_sizes, int n_in,
                              void* d_out, int out_size, void* d_ws, size_t ws_size,
                              hipStream_t stream) {
    (void)in_sizes; (void)n_in; (void)out_size; (void)ws_size;
    const float* xA    = (const float*)d_in[0];
    const float* xB    = (const float*)d_in[1];
    const float* Wk    = (const float*)d_in[2];
    const float* bk    = (const float*)d_in[3];
    const float* Wq    = (const float*)d_in[4];
    const float* bq    = (const float*)d_in[5];
    const float* Wv    = (const float*)d_in[6];
    const float* bv    = (const float*)d_in[7];
    const float* a_rel = (const float*)d_in[8];
    const float* m_rel = (const float*)d_in[9];
    const float* p_rel = (const float*)d_in[10];
    const float* Wa    = (const float*)d_in[11];
    const float* ba    = (const float*)d_in[12];
    const float* skip  = (const float*)d_in[13];
    const float* lin_w = (const float*)d_in[14];
    const float* lin_b = (const float*)d_in[15];
    const int*   ei[3] = {(const int*)d_in[16], (const int*)d_in[17], (const int*)d_in[18]};

    const size_t NF = (size_t)NNODE * FDIM;
    float* ws = (float*)d_ws;
    float* x[2]   = {ws,           ws + NF};
    float* q[2]   = {ws + 2 * NF,  ws + 3 * NF};
    float* k[2]   = {ws + 4 * NF,  ws + 5 * NF};
    float* v[2]   = {ws + 6 * NF,  ws + 7 * NF};
    float* kr     =  ws + 8 * NF;
    float* vr     =  ws + 9 * NF;
    float* agg[2] = {ws + 10 * NF, ws + 11 * NF};
    float* score  =  ws + 12 * NF;
    float* den    =  score + NEDGE;
    unsigned* mxb = (unsigned*)(den + NNODE);
    float* pooled = (float*)(mxb + NNODE);

    hipMemcpyAsync(x[0], xA, NF * sizeof(float), hipMemcpyDeviceToDevice, stream);
    hipMemcpyAsync(x[1], xB, NF * sizeof(float), hipMemcpyDeviceToDevice, stream);

    const dim3 gGrid(NNODE / 128), gBlk(256);
    const int RS[3] = {0, 1, 0}, RD[3] = {1, 0, 0};   // REL = [(0,1),(1,0),(0,0)]

    for (int l = 0; l < 3; ++l) {
        for (int t = 0; t < 2; ++t) {
            size_t wo = (size_t)(l * 2 + t) * 4096, bo = (size_t)(l * 2 + t) * 64;
            wmma_gemm64<<<gGrid, gBlk, 0, stream>>>(x[t], Wk + wo, bk + bo, k[t], nullptr, nullptr, 0);
            wmma_gemm64<<<gGrid, gBlk, 0, stream>>>(x[t], Wq + wo, bq + bo, q[t], nullptr, nullptr, 0);
            wmma_gemm64<<<gGrid, gBlk, 0, stream>>>(x[t], Wv + wo, bv + bo, v[t], nullptr, nullptr, 0);
        }
        hipMemsetAsync(agg[0], 0, NF * sizeof(float), stream);
        hipMemsetAsync(agg[1], 0, NF * sizeof(float), stream);
        for (int r = 0; r < 3; ++r) {
            int s = RS[r], d = RD[r];
            size_t ro = (size_t)(l * 3 + r) * 4096;
            wmma_gemm64<<<gGrid, gBlk, 0, stream>>>(k[s], a_rel + ro, nullptr, kr, nullptr, nullptr, 0);
            wmma_gemm64<<<gGrid, gBlk, 0, stream>>>(v[s], m_rel + ro, nullptr, vr, nullptr, nullptr, 0);
            init_mx_kernel<<<NNODE / 256, 256, 0, stream>>>(mxb, NNODE);
            hipMemsetAsync(den, 0, NNODE * sizeof(float), stream);
            score_max_kernel<<<NEDGE / 8, 256, 0, stream>>>(q[d], kr, ei[r], ei[r] + NEDGE,
                                                            p_rel + (l * 3 + r), score, mxb);
            exp_sum_kernel<<<(NEDGE + 255) / 256, 256, 0, stream>>>(score, ei[r] + NEDGE, mxb, den);
            scatter_kernel<<<NEDGE / 8, 256, 0, stream>>>(score, den, vr, ei[r], ei[r] + NEDGE, agg[d]);
        }
        for (int t = 0; t < 2; ++t) {
            size_t wo = (size_t)(l * 2 + t) * 4096, bo = (size_t)(l * 2 + t) * 64;
            wmma_gemm64<<<gGrid, gBlk, 0, stream>>>(agg[t], Wa + wo, ba + bo, x[t],
                                                    x[t], skip + (l * 2 + t), 1);
        }
    }

    hipMemsetAsync(pooled, 0, 64 * sizeof(float), stream);
    pool_kernel<<<512, 256, 0, stream>>>(x[0], pooled);
    pool_kernel<<<512, 256, 0, stream>>>(x[1], pooled);
    final_kernel<<<1, 32, 0, stream>>>(pooled, lin_w, lin_b, (float*)d_out);
}
